// NCA_4956392259657
// MI455X (gfx1250) — matrix operand order repaired and checked by reading, not compiled, not run on hardware
//
#include <hip/hip_runtime.h>

typedef __attribute__((ext_vector_type(16))) _Float16 v16h;
typedef __attribute__((ext_vector_type(4)))  _Float16 v4h;
typedef __attribute__((ext_vector_type(8)))  float    v8f;

namespace {
constexpr int Bn  = 16;   // batch
constexpr int Cn  = 8;    // channels
constexpr int Hn  = 512;
constexpr int Wn  = 512;
constexpr int Fn  = 5;    // stencils
constexpr int HID = 32;
constexpr int XE  = Wn + 2;         // extended columns (wrap halo baked in)
constexpr int NSL = 4;              // WMMA K-slices: KP = 128 = 8 ch * 16 slots
// K slot within a channel: k' = dx*4 + dy ; dy in 0..2 (+1 pad), dx in 0..2 (+1 pad)
}

// ---------------------------------------------------------------------------
// Prep: fold w1[8,32] @ w0[32,40] @ filters[5,3,3] -> Weff[8][128] (f16) in the
// wave32 WMMA B-fragment layout: wsB[t][lane][e] = Weff[n=lane&15][K], with
// K = 32t + 16*(lane>=16) + e.  K decode: ci = K>>4, dx = (K&15)>>2, dy = K&3;
// pad slots (dx==3 || dy==3) and columns n>=8 are zero.
// ---------------------------------------------------------------------------
__global__ void nca_prep(const float* __restrict__ filters,  // [5,3,3]
                         const float* __restrict__ w0,       // [32,40]
                         const float* __restrict__ w1,       // [8,32]
                         _Float16* __restrict__ wsB)         // [4][32][16]
{
    __shared__ float Wf[Cn * Cn * Fn];  // fused w1@w0 : [8][40]
    for (int i = threadIdx.x; i < Cn * Cn * Fn; i += blockDim.x) {
        int n = i / (Cn * Fn), g = i % (Cn * Fn);
        float s = 0.f;
        for (int h = 0; h < HID; ++h)
            s += w1[n * HID + h] * w0[h * (Cn * Fn) + g];
        Wf[i] = s;
    }
    __syncthreads();

    for (int i = threadIdx.x; i < NSL * 32 * 16; i += blockDim.x) {
        int t    = i >> 9;          // WMMA slice 0..3
        int lane = (i >> 4) & 31;
        int e    = i & 15;
        int n    = lane & 15;       // output channel (column)
        int K    = 32 * t + 16 * (lane >> 4) + e;
        int ci   = K >> 4;          // input channel
        int dx   = (K & 15) >> 2;   // 0..2 = x-1,x,x+1 ; 3 = pad
        int dy   = K & 3;           // 0..2 = y-1,y,y+1 ; 3 = pad
        float v  = 0.f;
        if (n < Cn && dx < 3 && dy < 3) {
            float s = 0.f;
            for (int f = 0; f < Fn; ++f)
                s += Wf[n * (Cn * Fn) + ci * Fn + f] * filters[f * 9 + dy * 3 + dx];
            v = s;
        }
        wsB[i] = (_Float16)v;
    }
}

// ---------------------------------------------------------------------------
// Main: one block per (b, y) row; 8 waves x 4 groups x 16 pixels = 512 px.
// Phase 1 (coalesced): stage "column triplets" in LDS: one 8-byte packet per
// (channel, extended column) = {row y-1, row y, row y+1, 0} as 4x f16.
// Phase 2: each lane assembles its A fragment from 4 adjacent-ulong pairs
// (ds_load_b64), runs 4x v_wmma_f32_16x16x32_f16, then masked residual+clamp.
// ---------------------------------------------------------------------------
__global__ void __launch_bounds__(256)
nca_step(const float* __restrict__ grid,
         const float* __restrict__ noise,
         const _Float16* __restrict__ wsB,
         float* __restrict__ out)
{
    __shared__ unsigned long long col4[Cn * XE + 1];  // 4113 * 8B = ~32.1 KB

    const int tid    = threadIdx.x;
    const int lane   = tid & 31;
    const int wave   = tid >> 5;
    const int lanehi = lane >> 4;
    const int m      = lane & 15;       // A row (pixel-in-group) & D column (channel)

    const int by = blockIdx.x;          // 0 .. B*H-1
    const int b  = by / Hn;
    const int y  = by % Hn;

    const int yw0 = (y == 0) ? Hn - 1 : y - 1;
    const int yw2 = (y == Hn - 1) ? 0 : y + 1;

    const size_t plane = (size_t)Hn * Wn;
    const float* gbase = grid  + (size_t)b * Cn * plane;
    const float* nbase = noise + (size_t)b * Cn * plane;
    float*       obase = out   + (size_t)b * Cn * plane;

    // Pull the next row's stream ahead (global_prefetch_b8)
    __builtin_prefetch(gbase + (size_t)yw2 * Wn + wave * 64, 0, 1);

    // ---- Phase 1: stage column triplets (fully coalesced, f32 -> packed f16)
    for (int i = tid; i < Cn * XE; i += 256) {
        int ci = i / XE;
        int xe = i - ci * XE;
        int xc = (xe == 0) ? (Wn - 1) : ((xe == XE - 1) ? 0 : (xe - 1));
        const float* gp = gbase + (size_t)ci * plane;
        float r0 = gp[(size_t)yw0 * Wn + xc];
        float r1 = gp[(size_t)y   * Wn + xc];
        float r2 = gp[(size_t)yw2 * Wn + xc];
        union { v4h h; unsigned long long u; } pk;
        pk.h[0] = (_Float16)r0;
        pk.h[1] = (_Float16)r1;
        pk.h[2] = (_Float16)r2;
        pk.h[3] = (_Float16)0.f;    // dy-pad slot (B row is zero anyway)
        col4[i] = pk.u;
    }
    if (tid == 0) col4[Cn * XE] = 0ull;   // tail packet read by ci=7, xe=512, lanehi=1
    __syncthreads();

    // ---- B fragments (uniform across the dispatch)
    const v16h* Bv = (const v16h*)wsB;
    v16h bf[NSL];
    #pragma unroll
    for (int t = 0; t < NSL; ++t) bf[t] = Bv[t * 32 + lane];

    // ---- Phase 2: 4 groups of 16 pixels per wave
    for (int g = 0; g < 4; ++g) {
        const int xg = (wave * 4 + g) * 16;   // group base x
        const int xe = xg + m + 1;            // extended-column index of this pixel

        // Slice t covers channel 2t + lanehi; each lane reads packets {x-1, x, x+1, x+2},
        // where the x+2 packet maps to zero B-rows (finite garbage is harmless).
        v8f c = {};
        #pragma unroll
        for (int t = 0; t < NSL; ++t) {
            const int i0 = (2 * t + lanehi) * XE + xe - 1;
            union { v16h h; unsigned long long u[4]; } au;
            au.u[0] = col4[i0];
            au.u[1] = col4[i0 + 1];
            au.u[2] = col4[i0 + 2];
            au.u[3] = col4[i0 + 3];
            c = __builtin_amdgcn_wmma_f32_16x16x32_f16(false, au.h, false, bf[t],
                                                       (short)0, c, false, false);
        }

        // Epilogue: lane's channel n = m (valid n<8); c[r] is pixel xg + r + 8*lanehi
        if (m < Cn) {
            const size_t rowoff = (size_t)m * plane + (size_t)y * Wn;
            const float* gch = gbase + rowoff;
            const float* nch = nbase + rowoff;
            float*       och = obase + rowoff;
            #pragma unroll
            for (int r = 0; r < 8; ++r) {
                const int p = xg + r + (lanehi << 3);
                const float gv  = gch[p];
                const float msk = (nch[p] < 0.5f) ? 1.0f : 0.0f;
                float res = gv + c[r] * msk;                 // DT = 1.0
                res = fminf(fmaxf(res, 0.0f), 1.0f);         // hard clamp
                och[p] = res;
            }
        }
    }
}

// ---------------------------------------------------------------------------
extern "C" void kernel_launch(void* const* d_in, const int* /*in_sizes*/, int /*n_in*/,
                              void* d_out, int /*out_size*/, void* d_ws, size_t /*ws_size*/,
                              hipStream_t stream)
{
    const float* grid    = (const float*)d_in[0];
    const float* noise   = (const float*)d_in[1];
    const float* filters = (const float*)d_in[2];
    const float* w0      = (const float*)d_in[3];
    const float* w1      = (const float*)d_in[4];
    float*       out     = (float*)d_out;
    _Float16*    wsB     = (_Float16*)d_ws;   // 4*32*16 f16 = 4 KB

    nca_prep<<<1, 128, 0, stream>>>(filters, w0, w1, wsB);
    nca_step<<<Bn * Hn, 256, 0, stream>>>(grid, noise, wsB, out);
}
